// CrystalLayer_16784732193213
// MI455X (gfx1250) — matrix operand order repaired and checked by reading, not compiled, not run on hardware
//
// CDNA5 (gfx1250) implementation of the CrystalLayer forward pass.
// Dense GEMMs: bf16 WMMA (v_wmma_f32_16x16x32_bf16) tile kernel, A tiles staged
// via global_load_async_to_lds_b128 (ASYNCcnt), fused epilogues.
// Attention: per-(batch,head) WMMA with spill-free shuffle softmax.
// Graph segment softmax: global float atomic max/add. Workspace ~621 MB.
#include <hip/hip_runtime.h>
#include <hip/hip_bf16.h>
#include <math.h>

typedef __attribute__((ext_vector_type(16))) __bf16 v16bf;
typedef __attribute__((ext_vector_type(8)))  float  v8f;

#define NN 16384
#define EE 196608
#define DD 256
#define BBATCH 256
#define SSEQ 64
#define HHEADS 8
#define DHEAD 32

static constexpr size_t ND  = (size_t)NN * DD;        // 4,194,304
static constexpr size_t ED  = (size_t)EE * DD;        // 50,331,648
static constexpr size_t ND2 = (size_t)NN * 2 * DD;    // 8,388,608

__device__ __constant__ const float kBnInv = 0.99999500003749968f; // 1/sqrt(1+1e-5)
__device__ __constant__ const float kLog2  = 0.6931471805599453f;

// ---------------- workspace layout (floats) ----------------
static constexpr size_t o_hA   = 0;
static constexpr size_t o_hB   = o_hA + ND;
static constexpr size_t o_hV   = o_hB + ND;
static constexpr size_t o_hU   = o_hV + ND;
static constexpr size_t o_tmp2 = o_hU + ND;          // ND2: nodeHid / t512 / fcHid
static constexpr size_t o_eA   = o_tmp2 + ND2;       // ED: hidC -> e_new
static constexpr size_t o_eB   = o_eA + ED;          // ED: e_C; later q/k/v/o
static constexpr size_t o_q    = o_eB;
static constexpr size_t o_k    = o_eB + ND;
static constexpr size_t o_v    = o_eB + 2 * ND;
static constexpr size_t o_o    = o_eB + 3 * ND;
static constexpr size_t o_m    = o_eB + ED;
static constexpr size_t o_s    = o_m + ND;
static constexpr size_t o_aggr = o_s + ND;
static constexpr size_t o_hLoc = o_aggr + ND;
static constexpr size_t o_hAtt = o_hLoc + ND;
static constexpr size_t o_tpre = o_hAtt + ND;
static constexpr size_t o_hNew = o_tpre + ND;
static constexpr size_t o_stats= o_hNew + ND;        // 2 floats

// ---------------- helpers ----------------
union FragU { v16bf v; uint4 q[2]; };

__device__ inline uint4 pack8bf(const float* f) {
  union { __bf16 h[8]; uint4 q; } u;
#pragma unroll
  for (int i = 0; i < 8; ++i) u.h[i] = (__bf16)f[i];
  return u.q;
}

// async 16B global -> LDS copy (ASYNCcnt tracked). dst is an LDS byte offset
// (ptrtoint of an addrspace(3) pointer), src a 64-bit global VA.
__device__ inline void async_copy16(unsigned lds_off, const void* src) {
  asm volatile("global_load_async_to_lds_b128 %0, %1, off"
               :: "v"(lds_off), "v"((unsigned long long)(uintptr_t)src)
               : "memory");
}
__device__ inline void wait_async0() {
  asm volatile("s_wait_asynccnt 0" ::: "memory");
}

// A fragment from a *fp32* LDS tile (converted in-register to bf16):
// lane m(0-15) half0: K{0..7,16..23}; half1: K{8..15,24..31}
__device__ inline v16bf load_frag_a_f32(const float* base, int ld, int lane) {
  const int m = lane & 15, h = lane >> 4;
  const float* p = base + (size_t)m * ld;
  float f[16];
  *(float4*)(f + 0)  = *(const float4*)(p + h * 8);
  *(float4*)(f + 4)  = *(const float4*)(p + h * 8 + 4);
  *(float4*)(f + 8)  = *(const float4*)(p + 16 + h * 8);
  *(float4*)(f + 12) = *(const float4*)(p + 16 + h * 8 + 4);
  union { v16bf v; __bf16 hh[16]; } u;
#pragma unroll
  for (int i = 0; i < 16; ++i) u.hh[i] = (__bf16)f[i];
  return u.v;
}
// A fragment from a bf16 LDS tile (attention probs)
__device__ inline v16bf load_frag_a(const __bf16* base, int ld, int lane) {
  const int m = lane & 15, h = lane >> 4;
  const uint4* p = (const uint4*)(base + (size_t)m * ld);
  FragU u; u.q[0] = p[h]; u.q[1] = p[2 + h];
  return u.v;
}
// B fragment (32x16): lane n(0-15)=column n, half0: K{0..7,16..23}; half1: K{8..15,24..31}
// (LDS tile stored transposed: [n][k], bf16)
__device__ inline v16bf load_frag_b(const __bf16* base, int ld, int lane) {
  const int n = lane & 15, h = lane >> 4;
  const uint4* p = (const uint4*)(base + (size_t)n * ld);
  FragU u; u.q[0] = p[h]; u.q[1] = p[2 + h];
  return u.v;
}

__device__ inline v8f wmma_bf16(v16bf a, v16bf b, v8f c) {
  return __builtin_amdgcn_wmma_f32_16x16x32_bf16(false, a, false, b, (short)0, c, false, false);
}

template <int ACT> __device__ inline float act_fn(float z) {
  if (ACT == 1) return z / (1.f + expf(-z));                                 // SiLU
  if (ACT == 2) return 0.5f * z * (1.f + erff(z * 0.70710678118654752f));    // exact GELU
  if (ACT == 3) return (z > 20.f ? z : log1pf(expf(z))) - kLog2;             // shifted softplus
  return z;
}

// ---------------- bf16 WMMA GEMM: C = epi(A[M,K] @ W[K,Nw] + bias) ----------------
// block tile 128x64, 8 waves of 32x32 (2x2 frags), K-step 32.
// A tile staged fp32 by async LDS copies; W tile transposed+converted to bf16.
template <int ACT, bool RES, bool BN>
__global__ __launch_bounds__(256)
void gemm_bf16_kernel(const float* __restrict__ A, const float* __restrict__ W,
                      const float* __restrict__ bias, const float* __restrict__ res,
                      const float* __restrict__ bng, const float* __restrict__ bnb,
                      float* __restrict__ C, int M, int K, int Nw) {
  __shared__ __align__(16) float  Asf[128 * 36];   // fp32 A tile, padded ld=36
  __shared__ __align__(16) __bf16 Ws[64 * 40];     // bf16 W^T tile, ld=40

  const int tid  = threadIdx.x;
  const int lane = tid & 31, wid = tid >> 5;
  const int wm = wid & 3, wn = wid >> 2;
  const int m0 = blockIdx.y * 128;
  const int n0 = blockIdx.x * 64;

  v8f acc[2][2];
#pragma unroll
  for (int i = 0; i < 2; ++i)
#pragma unroll
    for (int j = 0; j < 2; ++j)
#pragma unroll
      for (int v = 0; v < 8; ++v) acc[i][j][v] = 0.f;

  const int rA = tid >> 1, hAf = (tid & 1) * 16;    // A stage: 16 floats / thread
  const int nW = tid & 63, kq = tid >> 6;           // W stage: 8 floats / thread

  for (int k0 = 0; k0 < K; k0 += 32) {
    // ---- async-stage A tile (128 x 32 fp32): 4x 16B copies per thread ----
    {
      const float* src = A + (size_t)(m0 + rA) * K + k0 + hAf;
      const unsigned dst = (unsigned)(uintptr_t)(Asf + rA * 36 + hAf);
#pragma unroll
      for (int j = 0; j < 4; ++j)
        async_copy16(dst + j * 16, src + j * 4);
    }
    // ---- stage W tile (32 x 64) transposed -> Ws[n][k] bf16 ----
    {
      float t[8];
#pragma unroll
      for (int j = 0; j < 8; ++j)
        t[j] = W[(size_t)(k0 + kq * 8 + j) * Nw + n0 + nW];
      *(uint4*)(Ws + nW * 40 + kq * 8) = pack8bf(t);
    }
    wait_async0();
    __syncthreads();

    v16bf a0 = load_frag_a_f32(Asf + (wm * 32)      * 36, 36, lane);
    v16bf a1 = load_frag_a_f32(Asf + (wm * 32 + 16) * 36, 36, lane);
    v16bf b0 = load_frag_b(Ws + (wn * 32)      * 40, 40, lane);
    v16bf b1 = load_frag_b(Ws + (wn * 32 + 16) * 40, 40, lane);
    acc[0][0] = wmma_bf16(a0, b0, acc[0][0]);
    acc[0][1] = wmma_bf16(a0, b1, acc[0][1]);
    acc[1][0] = wmma_bf16(a1, b0, acc[1][0]);
    acc[1][1] = wmma_bf16(a1, b1, acc[1][1]);
    __syncthreads();
  }

  // ---- epilogue: C layout lane L, vgpr v -> row = v + 8*(L>=16), col = L%16 ----
  const int l16 = lane & 15, hi = lane >> 4;
#pragma unroll
  for (int i = 0; i < 2; ++i) {
#pragma unroll
    for (int j = 0; j < 2; ++j) {
      const int col = n0 + wn * 32 + j * 16 + l16;
      const float bsv = bias ? bias[col] : 0.f;
      const float gsc = BN ? (kBnInv * bng[col]) : 0.f;
      const float gsh = BN ? bnb[col] : 0.f;
#pragma unroll
      for (int v = 0; v < 8; ++v) {
        const int row = m0 + wm * 32 + i * 16 + v + hi * 8;
        float val = acc[i][j][v] + bsv;
        val = act_fn<ACT>(val);
        if (RES) val += res[(size_t)row * Nw + col];
        if (BN)  val = val * gsc + gsh;
        C[(size_t)row * Nw + col] = val;
      }
    }
  }
}

// ---------------- attention: per (b,h) block, WMMA scores + softmax + WMMA PV ----------------
__global__ __launch_bounds__(256)
void attn_kernel(const float* __restrict__ qb, const float* __restrict__ kb,
                 const float* __restrict__ vb, float* __restrict__ ob) {
  __shared__ __align__(16) __bf16 Qs[64 * 40];
  __shared__ __align__(16) __bf16 Ks[64 * 40];
  __shared__ __align__(16) __bf16 Vs[32 * 72];   // transposed: [dh][s]
  __shared__ __align__(16) __bf16 Ps[64 * 72];   // probs, ld=72
  __shared__ __align__(16) float  Ss[64 * 68];   // scores fp32, ld=68

  const int bh = blockIdx.x;
  const int b = bh / HHEADS, h = bh % HHEADS;
  const int tid = threadIdx.x, lane = tid & 31, wid = tid >> 5;
  const size_t base = (size_t)b * SSEQ * DD + h * DHEAD;

  // stage Q,K: thread t -> row s=t/4, 8 dh elems
  {
    const int s = tid >> 2, dh0 = (tid & 3) * 8;
    float t[8];
#pragma unroll
    for (int j = 0; j < 2; ++j) ((float4*)t)[j] = ((const float4*)(qb + base + (size_t)s * DD + dh0))[j];
    *(uint4*)(Qs + s * 40 + dh0) = pack8bf(t);
#pragma unroll
    for (int j = 0; j < 2; ++j) ((float4*)t)[j] = ((const float4*)(kb + base + (size_t)s * DD + dh0))[j];
    *(uint4*)(Ks + s * 40 + dh0) = pack8bf(t);
  }
  // stage V transposed: thread t -> dh = t%32, 8 seq elems
  {
    const int dh = tid & 31, s0 = (tid >> 5) * 8;
    float t[8];
#pragma unroll
    for (int j = 0; j < 8; ++j) t[j] = vb[base + (size_t)(s0 + j) * DD + dh];
    *(uint4*)(Vs + dh * 72 + s0) = pack8bf(t);
  }
  __syncthreads();

  // scores = Q @ K^T / sqrt(DH) : M=64,N=64,K=32 -> 16 tiles, 2 per wave
  {
    const int tm = wid & 3, tnb = (wid >> 2) * 2;
    v16bf a = load_frag_a(Qs + tm * 16 * 40, 40, lane);
    const float scale = 0.17677669529663687f; // 1/sqrt(32)
#pragma unroll
    for (int jj = 0; jj < 2; ++jj) {
      const int tn = tnb + jj;
      v16bf bq = load_frag_b(Ks + tn * 16 * 40, 40, lane);
      v8f c;
#pragma unroll
      for (int v = 0; v < 8; ++v) c[v] = 0.f;
      c = wmma_bf16(a, bq, c);
      const int l16 = lane & 15, hi = lane >> 4;
#pragma unroll
      for (int v = 0; v < 8; ++v)
        Ss[(tm * 16 + v + hi * 8) * 68 + tn * 16 + l16] = c[v] * scale;
    }
  }
  __syncthreads();

  // row softmax: 4 threads per row, 16 statically-indexed elems each (no spill),
  // cross-lane max/sum via shuffles (4 threads of a row share a wave).
  {
    const int row = tid >> 2, q = tid & 3;
    const float* srow = Ss + row * 68 + q * 16;
    float ev[16];
    float mx = -INFINITY;
#pragma unroll
    for (int c = 0; c < 16; ++c) mx = fmaxf(mx, srow[c]);
    mx = fmaxf(mx, __shfl_xor(mx, 1));
    mx = fmaxf(mx, __shfl_xor(mx, 2));
    float sum = 0.f;
#pragma unroll
    for (int c = 0; c < 16; ++c) { ev[c] = expf(srow[c] - mx); sum += ev[c]; }
    sum += __shfl_xor(sum, 1);
    sum += __shfl_xor(sum, 2);
    const float inv = 1.f / sum;
#pragma unroll
    for (int c = 0; c < 16; ++c) ev[c] *= inv;
    *(uint4*)(Ps + row * 72 + q * 16)     = pack8bf(ev);
    *(uint4*)(Ps + row * 72 + q * 16 + 8) = pack8bf(ev + 8);
  }
  __syncthreads();

  // out = P[64,64] @ V[64,32] : M=64,N=32,K=64 -> 8 tiles, 1 per wave
  {
    const int tm = wid & 3, tn = wid >> 2;
    v8f c;
#pragma unroll
    for (int v = 0; v < 8; ++v) c[v] = 0.f;
#pragma unroll
    for (int ks = 0; ks < 64; ks += 32) {
      v16bf a  = load_frag_a(Ps + tm * 16 * 72 + ks, 72, lane);
      v16bf bv = load_frag_b(Vs + tn * 16 * 72 + ks, 72, lane);
      c = wmma_bf16(a, bv, c);
    }
    const int l16 = lane & 15, hi = lane >> 4;
#pragma unroll
    for (int v = 0; v < 8; ++v) {
      const int srow = tm * 16 + v + hi * 8;
      const int dh   = tn * 16 + l16;
      ob[base + (size_t)srow * DD + dh] = c[v];
    }
  }
}

// ---------------- elementwise / graph kernels ----------------
__global__ void k_fill(float* p, float v, size_t n) {
  size_t i = blockIdx.x * (size_t)blockDim.x + threadIdx.x;
  size_t stride = (size_t)gridDim.x * blockDim.x;
  for (; i < n; i += stride) p[i] = v;
}

// e_new = edge_attr + relu(bn_e(hA[row]+hB[col]+eC)); segment max into m[col]
__global__ void k_edge_score(const float* __restrict__ eAttr, const float* __restrict__ eC,
                             const float* __restrict__ hA, const float* __restrict__ hB,
                             const int* __restrict__ ei, const float* __restrict__ g,
                             const float* __restrict__ bbn, float* __restrict__ eNew,
                             float* __restrict__ m) {
  size_t idx = (blockIdx.x * (size_t)blockDim.x + threadIdx.x) * 4;
  if (idx >= ED) return;
  const int e = (int)(idx / DD), d = (int)(idx % DD);
  const int r = ei[e], c = ei[EE + e];
#pragma unroll
  for (int u = 0; u < 4; ++u) {
    const int dd = d + u;
    float sc = hA[(size_t)r * DD + dd] + hB[(size_t)c * DD + dd] + eC[idx + u];
    sc = sc * (kBnInv * g[dd]) + bbn[dd];
    sc = fmaxf(sc, 0.f);
    const float en = eAttr[idx + u] + sc;
    eNew[idx + u] = en;
    __hip_atomic_fetch_max(&m[(size_t)c * DD + dd], en,
                           __ATOMIC_RELAXED, __HIP_MEMORY_SCOPE_AGENT);
  }
}

__global__ void k_edge_expsum(const float* __restrict__ eNew, const int* __restrict__ ei,
                              const float* __restrict__ m, float* __restrict__ s) {
  size_t idx = (blockIdx.x * (size_t)blockDim.x + threadIdx.x) * 4;
  if (idx >= ED) return;
  const int e = (int)(idx / DD), d = (int)(idx % DD);
  const int c = ei[EE + e];
#pragma unroll
  for (int u = 0; u < 4; ++u) {
    const int dd = d + u;
    atomicAdd(&s[(size_t)c * DD + dd], expf(eNew[idx + u] - m[(size_t)c * DD + dd]));
  }
}

__global__ void k_edge_msg(const float* __restrict__ eNew, const int* __restrict__ ei,
                           const float* __restrict__ m, const float* __restrict__ s,
                           const float* __restrict__ hV, float* __restrict__ aggr) {
  size_t idx = (blockIdx.x * (size_t)blockDim.x + threadIdx.x) * 4;
  if (idx >= ED) return;
  const int e = (int)(idx / DD), d = (int)(idx % DD);
  const int c = ei[EE + e];
#pragma unroll
  for (int u = 0; u < 4; ++u) {
    const int dd = d + u;
    const size_t ci = (size_t)c * DD + dd;
    const float en = eNew[idx + u];
    const float gated = expf(en - m[ci]) / (s[ci] + 1e-16f);
    atomicAdd(&aggr[ci], gated * (hV[ci] + en));
  }
}

// h_local = bn1l(silu(x + bn_h(hU + aggr)))
__global__ void k_local(const float* __restrict__ x, const float* __restrict__ hU,
                        const float* __restrict__ aggr, const float* __restrict__ gh,
                        const float* __restrict__ bh, const float* __restrict__ g1,
                        const float* __restrict__ b1, float* __restrict__ hLoc) {
  size_t i = blockIdx.x * (size_t)blockDim.x + threadIdx.x;
  if (i >= ND) return;
  const int d = (int)(i % DD);
  float t = hU[i] + aggr[i];
  t = t * (kBnInv * gh[d]) + bh[d];
  float v = x[i] + t;
  v = v / (1.f + expf(-v));
  hLoc[i] = v * (kBnInv * g1[d]) + b1[d];
}

__global__ void k_reduce(const float* __restrict__ p, float* __restrict__ stats, size_t n) {
  __shared__ float s1[256], s2[256];
  size_t i = blockIdx.x * (size_t)blockDim.x + threadIdx.x;
  const size_t stride = (size_t)gridDim.x * blockDim.x;
  float a = 0.f, b = 0.f;
  for (; i < n; i += stride) { const float v = p[i]; a += v; b += v * v; }
  s1[threadIdx.x] = a; s2[threadIdx.x] = b;
  __syncthreads();
  for (int off = 128; off > 0; off >>= 1) {
    if (threadIdx.x < off) { s1[threadIdx.x] += s1[threadIdx.x + off]; s2[threadIdx.x] += s2[threadIdx.x + off]; }
    __syncthreads();
  }
  if (threadIdx.x == 0) { atomicAdd(&stats[0], s1[0]); atomicAdd(&stats[1], s2[0]); }
}

// h_new = 0.5*(h_local + graph_ln(tpre)) with global mean/var
__global__ void k_ln_fuse(const float* __restrict__ tpre, const float* __restrict__ hLoc,
                          const float* __restrict__ stats, const float* __restrict__ g,
                          const float* __restrict__ bln, float* __restrict__ hNew) {
  size_t i = blockIdx.x * (size_t)blockDim.x + threadIdx.x;
  if (i >= ND) return;
  const float invn = 1.f / (float)ND;
  const float mu = stats[0] * invn;
  const float var = stats[1] * invn - mu * mu;
  const float rs = rsqrtf(var + 1e-5f);
  const int d = (int)(i % DD);
  const float ha = (tpre[i] - mu) * rs * g[d] + bln[d];
  hNew[i] = 0.5f * (hLoc[i] + ha);
}

// ---------------- launch ----------------
static inline void gemm_launch_dims(int M, int Nw, dim3& g) { g = dim3(Nw / 64, M / 128, 1); }

extern "C" void kernel_launch(void* const* d_in, const int* in_sizes, int n_in,
                              void* d_out, int out_size, void* d_ws, size_t ws_size,
                              hipStream_t stream) {
  const float* x      = (const float*)d_in[0];
  const int*   ei     = (const int*)  d_in[1];
  const float* eAttr  = (const float*)d_in[2];
  const float* nw1    = (const float*)d_in[3];
  const float* nb1    = (const float*)d_in[4];
  const float* nw2    = (const float*)d_in[5];
  const float* nb2    = (const float*)d_in[6];
  const float* cw1    = (const float*)d_in[7];
  const float* cb1    = (const float*)d_in[8];
  const float* cw2    = (const float*)d_in[9];
  const float* cb2    = (const float*)d_in[10];
  const float* bn_e_g = (const float*)d_in[11];
  const float* bn_e_b = (const float*)d_in[12];
  const float* bn_h_g = (const float*)d_in[13];
  const float* bn_h_b = (const float*)d_in[14];
  const float* n1l_g  = (const float*)d_in[15];
  const float* n1l_b  = (const float*)d_in[16];
  const float* n1a_g  = (const float*)d_in[17];
  const float* n1a_b  = (const float*)d_in[18];
  const float* n2_g   = (const float*)d_in[19];
  const float* n2_b   = (const float*)d_in[20];
  const float* wq     = (const float*)d_in[21];
  const float* bq     = (const float*)d_in[22];
  const float* wk     = (const float*)d_in[23];
  const float* bk     = (const float*)d_in[24];
  const float* wv     = (const float*)d_in[25];
  const float* bv     = (const float*)d_in[26];
  const float* wo     = (const float*)d_in[27];
  const float* bo     = (const float*)d_in[28];
  const float* affw1  = (const float*)d_in[29];
  const float* affb1  = (const float*)d_in[30];
  const float* affw2  = (const float*)d_in[31];
  const float* affb2  = (const float*)d_in[32];
  const float* fn_g   = (const float*)d_in[33];
  const float* fn_b   = (const float*)d_in[34];
  const float* fcw1   = (const float*)d_in[35];
  const float* fcb1   = (const float*)d_in[36];
  const float* fcw2   = (const float*)d_in[37];
  const float* fcb2   = (const float*)d_in[38];

  float* ws  = (float*)d_ws;
  float* out = (float*)d_out;

  const dim3 blk(256, 1, 1);
  dim3 g;

  // init segment buffers + stats
  k_fill<<<4096, blk, 0, stream>>>(ws + o_m, -INFINITY, ND);
  k_fill<<<8192, blk, 0, stream>>>(ws + o_s, 0.f, 2 * ND);  // s and aggr (contiguous)
  k_fill<<<1, 32, 0, stream>>>(ws + o_stats, 0.f, 2);

  // ---- node MLPs (A,B,V,U) ----
  for (int kk = 0; kk < 4; ++kk) {
    gemm_launch_dims(NN, DD, g);
    gemm_bf16_kernel<1, false, false><<<g, blk, 0, stream>>>(
        x, nw1 + (size_t)kk * DD * DD, nb1 + kk * DD, nullptr, nullptr, nullptr,
        ws + o_tmp2, NN, DD, DD);
    gemm_bf16_kernel<0, false, false><<<g, blk, 0, stream>>>(
        ws + o_tmp2, nw2 + (size_t)kk * DD * DD, nb2 + kk * DD, nullptr, nullptr, nullptr,
        ws + o_hA + (size_t)kk * ND, NN, DD, DD);
  }

  // ---- edge MLP: e_C ----
  gemm_launch_dims(EE, DD, g);
  gemm_bf16_kernel<1, false, false><<<g, blk, 0, stream>>>(
      eAttr, cw1, cb1, nullptr, nullptr, nullptr, ws + o_eA, EE, DD, DD);
  gemm_bf16_kernel<0, false, false><<<g, blk, 0, stream>>>(
      ws + o_eA, cw2, cb2, nullptr, nullptr, nullptr, ws + o_eB, EE, DD, DD);

  // ---- gated segment softmax over edges ----
  const int eb = (int)(ED / 4 / 256);
  k_edge_score<<<eb, blk, 0, stream>>>(eAttr, ws + o_eB, ws + o_hA, ws + o_hB, ei,
                                       bn_e_g, bn_e_b, ws + o_eA, ws + o_m);
  k_edge_expsum<<<eb, blk, 0, stream>>>(ws + o_eA, ei, ws + o_m, ws + o_s);
  k_edge_msg<<<eb, blk, 0, stream>>>(ws + o_eA, ei, ws + o_m, ws + o_s,
                                     ws + o_hV, ws + o_aggr);
  k_local<<<(int)(ND / 256), blk, 0, stream>>>(x, ws + o_hU, ws + o_aggr,
                                               bn_h_g, bn_h_b, n1l_g, n1l_b, ws + o_hLoc);

  // ---- transformer branch ----
  gemm_launch_dims(NN, DD, g);
  gemm_bf16_kernel<0, false, false><<<g, blk, 0, stream>>>(
      x, wq, bq, nullptr, nullptr, nullptr, ws + o_q, NN, DD, DD);
  gemm_bf16_kernel<0, false, false><<<g, blk, 0, stream>>>(
      x, wk, bk, nullptr, nullptr, nullptr, ws + o_k, NN, DD, DD);
  gemm_bf16_kernel<0, false, false><<<g, blk, 0, stream>>>(
      x, wv, bv, nullptr, nullptr, nullptr, ws + o_v, NN, DD, DD);

  attn_kernel<<<BBATCH * HHEADS, blk, 0, stream>>>(ws + o_q, ws + o_k, ws + o_v, ws + o_o);

  gemm_bf16_kernel<0, true, true><<<g, blk, 0, stream>>>(
      ws + o_o, wo, bo, x, n1a_g, n1a_b, ws + o_hAtt, NN, DD, DD);

  gemm_launch_dims(NN, 2 * DD, g);
  gemm_bf16_kernel<2, false, false><<<g, blk, 0, stream>>>(
      ws + o_hAtt, affw1, affb1, nullptr, nullptr, nullptr, ws + o_tmp2, NN, DD, 2 * DD);
  gemm_launch_dims(NN, DD, g);
  gemm_bf16_kernel<0, true, false><<<g, blk, 0, stream>>>(
      ws + o_tmp2, affw2, affb2, x, nullptr, nullptr, ws + o_tpre, NN, 2 * DD, DD);

  // graph LayerNorm over all N*D elements, fused with branch averaging
  k_reduce<<<1024, blk, 0, stream>>>(ws + o_tpre, ws + o_stats, ND);
  k_ln_fuse<<<(int)(ND / 256), blk, 0, stream>>>(ws + o_tpre, ws + o_hLoc,
                                                 ws + o_stats, fn_g, fn_b, ws + o_hNew);

  // ---- final FF block + norm2 -> d_out ----
  gemm_launch_dims(NN, 2 * DD, g);
  gemm_bf16_kernel<3, false, false><<<g, blk, 0, stream>>>(
      ws + o_hNew, fcw1, fcb1, nullptr, nullptr, nullptr, ws + o_tmp2, NN, DD, 2 * DD);
  gemm_launch_dims(NN, DD, g);
  gemm_bf16_kernel<0, true, true><<<g, blk, 0, stream>>>(
      ws + o_tmp2, fcw2, fcb2, ws + o_hNew, n2_g, n2_b, out, NN, 2 * DD, DD);

  (void)in_sizes; (void)n_in; (void)out_size; (void)ws_size;
}
